// AutoregU_23244363005996
// MI455X (gfx1250) — matrix operand order repaired and checked by reading, not compiled, not run on hardware
//
#include <hip/hip_runtime.h>
#include <hip/hip_bf16.h>

#define T_STEPS 12
#define N_NODESC 50000
#define N_EDGESC 800000
#define IN_FC 11
#define H_DIMC 128
#define OUT_FC 3
#define NPAD 50048      /* 391 * 128 */
#define KDIM 256

typedef __attribute__((ext_vector_type(16))) __bf16 v16bf;
typedef __attribute__((ext_vector_type(8)))  float  v8f;

union FragAB { v16bf v; uint4 u[2]; };
union FragC  { v8f   v; float f[8]; };

#if __has_builtin(__builtin_amdgcn_s_wait_asynccnt)
#define WAIT_ASYNC(n) __builtin_amdgcn_s_wait_asynccnt(n)
#else
#define WAIT_ASYNC(n) asm volatile("s_wait_asynccnt %0" ::"i"(n) : "memory")
#endif

static __device__ __forceinline__ unsigned lds_addr_of(const void* p) {
  // numeric LDS byte offset of a __shared__ pointer (addrspace(3) ptrtoint)
  return (unsigned)(size_t)(__attribute__((address_space(3))) const void*)p;
}

static __device__ __forceinline__ unsigned short f2bf(float x) {
  unsigned int u = __float_as_uint(x);
  u += 0x7FFFu + ((u >> 16) & 1u);   // round-to-nearest-even
  return (unsigned short)(u >> 16);
}

static __device__ __forceinline__ float sigmoidf(float x) {
  return 1.0f / (1.0f + __expf(-x));
}

// ---------------- edge-weight precompute ----------------
__global__ void k_deg(const int* __restrict__ ei, float* __restrict__ deg) {
  int e = blockIdx.x * blockDim.x + threadIdx.x;
  if (e >= N_EDGESC) return;
  int r = ei[e], c = ei[N_EDGESC + e];
  if (r != c) atomicAdd(&deg[r], 1.0f);
}

__global__ void k_dinv(const float* __restrict__ deg, float* __restrict__ dinv) {
  int n = blockIdx.x * blockDim.x + threadIdx.x;
  if (n >= N_NODESC) return;
  float d = deg[n];
  dinv[n] = (d > 0.0f) ? rsqrtf(fmaxf(d, 1e-12f)) : 0.0f;
}

__global__ void k_edgew(const int* __restrict__ ei, const float* __restrict__ dinv,
                        float* __restrict__ w) {
  int e = blockIdx.x * blockDim.x + threadIdx.x;
  if (e >= N_EDGESC) return;
  int r = ei[e], c = ei[N_EDGESC + e];
  w[e] = (r != c) ? -dinv[r] * dinv[c] : 0.0f;
}

// ---------------- weight prep: Bt[f][k] = W[k>>7][k&127][f] in bf16 --------
__global__ void k_prep_w(const float* __restrict__ W, unsigned short* __restrict__ Bt) {
  int idx = blockIdx.x * blockDim.x + threadIdx.x;
  if (idx >= 128 * 256) return;
  int f = idx >> 8;
  int k = idx & 255;
  Bt[f * 256 + k] = f2bf(W[(k >> 7) * 16384 + (k & 127) * 128 + f]);
}

// ---------------- x modification (autoregressive feedback) ------------------
__global__ void k_xmod(const float* __restrict__ X, const float* __restrict__ uprev,
                       float* __restrict__ xb, int t) {
  int n = blockIdx.x * blockDim.x + threadIdx.x;
  if (n >= N_NODESC) return;
  const float* xt = X + ((size_t)t * N_NODESC + n) * IN_FC;
  float x[12];
#pragma unroll
  for (int i = 0; i < 11; ++i) x[i] = xt[i];
  x[11] = 0.0f;
  if (t > 0) {
    const float* xp = X + ((size_t)(t - 1) * N_NODESC + n) * IN_FC;
    const float* up = uprev + (size_t)n * 3;
    float dt = xt[6] - xp[6];
#pragma unroll
    for (int j = 0; j < 3; ++j) {
      x[3 + j] = up[j];
      x[8 + j] = (up[j] - xp[3 + j]) / dt;
    }
  }
  float* o = xb + (size_t)n * 12;
#pragma unroll
  for (int i = 0; i < 12; ++i) o[i] = x[i];
}

// ---------------- lhat scatter: 11-feature (x) ------------------------------
__global__ void k_scatter_x(const int* __restrict__ ei, const float* __restrict__ w,
                            const float* __restrict__ xb, float* __restrict__ lx) {
  int e = blockIdx.x * blockDim.x + threadIdx.x;
  if (e >= N_EDGESC) return;
  float we = w[e];
  if (we == 0.0f) return;
  int r = ei[e], c = ei[N_EDGESC + e];
  const float* s = xb + (size_t)c * 12;
  float* d = lx + (size_t)r * 12;
#pragma unroll
  for (int i = 0; i < 11; ++i) atomicAdd(&d[i], we * s[i]);
}

// ---------------- lhat scatter: 128-feature (h / h*r) -----------------------
__global__ void k_scatter128(const int* __restrict__ ei, const float* __restrict__ w,
                             const float* __restrict__ src, float* __restrict__ dst) {
  int tid = blockIdx.x * blockDim.x + threadIdx.x;   // E*32 threads
  if (tid >= N_EDGESC * 32) return;
  int e = tid >> 5;
  int c = (tid & 31) << 2;
  float we = w[e];
  if (we == 0.0f) return;
  int r = ei[e], co = ei[N_EDGESC + e];
  const float4 hv = *reinterpret_cast<const float4*>(src + (size_t)co * 128 + c);
  float* d = dst + (size_t)r * 128 + c;
  atomicAdd(d + 0, we * hv.x);
  atomicAdd(d + 1, we * hv.y);
  atomicAdd(d + 2, we * hv.z);
  atomicAdd(d + 3, we * hv.w);
}

// ---------------- x-side contributions + biases for all 3 gates -------------
__global__ void k_xpart(const float* __restrict__ xb, const float* __restrict__ lx,
                        const float* __restrict__ Wxz, const float* __restrict__ Wxr,
                        const float* __restrict__ Wxh,
                        const float* __restrict__ bxz, const float* __restrict__ bhz,
                        const float* __restrict__ bxr, const float* __restrict__ bhr,
                        const float* __restrict__ bxh, const float* __restrict__ bhh,
                        float* __restrict__ az, float* __restrict__ ar,
                        float* __restrict__ ah) {
  int idx = blockIdx.x * blockDim.x + threadIdx.x;
  if (idx >= N_NODESC * 128) return;
  int n = idx >> 7, f = idx & 127;
  float sz = bxz[f] + bhz[f];
  float sr = bxr[f] + bhr[f];
  float sh = bxh[f] + bhh[f];
  const float* xr = xb + (size_t)n * 12;
  const float* lr = lx + (size_t)n * 12;
#pragma unroll
  for (int i = 0; i < 11; ++i) {
    float xv = xr[i], lv = lr[i];
    sz += xv * Wxz[i * 128 + f] + lv * Wxz[1408 + i * 128 + f];
    sr += xv * Wxr[i * 128 + f] + lv * Wxr[1408 + i * 128 + f];
    sh += xv * Wxh[i * 128 + f] + lv * Wxh[1408 + i * 128 + f];
  }
  az[idx] = sz; ar[idx] = sr; ah[idx] = sh;
}

// ---------------- bf16 packing ---------------------------------------------
__global__ void k_pack2(const float* __restrict__ a, const float* __restrict__ b,
                        unsigned short* __restrict__ ab) {
  int idx = blockIdx.x * blockDim.x + threadIdx.x;
  if (idx >= N_NODESC * 128) return;
  int n = idx >> 7, f = idx & 127;
  ab[(size_t)n * 256 + f] = f2bf(a[idx]);
  ab[(size_t)n * 256 + 128 + f] = f2bf(b[idx]);
}

__global__ void k_packhi(const float* __restrict__ b, unsigned short* __restrict__ ab) {
  int idx = blockIdx.x * blockDim.x + threadIdx.x;
  if (idx >= N_NODESC * 128) return;
  int n = idx >> 7, f = idx & 127;
  ab[(size_t)n * 256 + 128 + f] = f2bf(b[idx]);
}

// ---------------- WMMA GEMM: C[NPAD,128] += A[NPAD,256](bf16) @ Bt^T --------
// Bt is [128,256] row-major (output-col major): Bt[f][k].
// A tiles are staged into LDS with GLOBAL_LOAD_ASYNC_TO_LDS_B128 (ASYNCcnt),
// double-buffered so chunk k+1 streams in while chunk k feeds the WMMAs.
__global__ void __launch_bounds__(256)
k_wmma_gemm(const unsigned short* __restrict__ A,
            const unsigned short* __restrict__ Bt,
            float* __restrict__ C) {
  __shared__ unsigned short Atile[2][128 * 32];   // 2 x 8 KB double buffer

  const int tid = threadIdx.x;
  const int wave = tid >> 5;
  const int lane = tid & 31;
  const int lo = lane & 15;
  const int hi = lane >> 4;
  const int blockRow = blockIdx.x * 128;
  const int mBase = blockRow + wave * 16 + hi * 8;   // C/D: M = vgpr + 8*hi

  const unsigned ldsA = lds_addr_of(&Atile[0][0]);

  // async-stage one 128x32 bf16 A chunk (512 x b128; 2 per thread) into buf
  auto stage = [&](int buf, int k0) {
    const unsigned lbase = ldsA + (unsigned)buf * 8192u;
#pragma unroll
    for (int j = 0; j < 2; ++j) {
      const int idx = tid + j * 256;            // 0..511
      const int rr = idx >> 2;                  // row in tile
      const int c4 = idx & 3;                   // b128 chunk within 32-k slice
      const unsigned short* g = A + (size_t)(blockRow + rr) * 256 + k0 + c4 * 8;
      const unsigned l = lbase + (unsigned)idx * 16u;
      asm volatile("global_load_async_to_lds_b128 %0, %1, off"
                   :: "v"(l), "v"(g) : "memory");
    }
  };

  // load accumulators (C): lane (hi,lo) owns rows mBase..mBase+7, col tile*16+lo
  FragC acc[8];
#pragma unroll
  for (int t = 0; t < 8; ++t) {
    const int col = t * 16 + lo;
#pragma unroll
    for (int v = 0; v < 8; ++v)
      acc[t].f[v] = C[(size_t)(mBase + v) * 128 + col];
  }

  stage(0, 0);   // prologue

  for (int k0 = 0; k0 < KDIM; k0 += 32) {
    const int cur = (k0 >> 5) & 1;
    const bool more = (k0 + 32) < KDIM;
    if (more) {
      stage(cur ^ 1, k0 + 32);                  // next chunk in flight
      __builtin_prefetch(&A[(size_t)blockRow * 256 + k0 + 64], 0, 1);
      WAIT_ASYNC(2);   // my 2 current-buffer loads done (in-order), next 2 pending
    } else {
      WAIT_ASYNC(0);
    }
    __syncthreads();   // publish staged buffer to all waves

    // A fragment (16x32 bf16): lane-half hi=0 -> K {0..7,16..23}, hi=1 -> {8..15,24..31}
    FragAB afrag;
    {
      const unsigned short* ap = &Atile[cur][(wave * 16 + lo) * 32 + hi * 8];
      afrag.u[0] = *reinterpret_cast<const uint4*>(ap);
      afrag.u[1] = *reinterpret_cast<const uint4*>(ap + 16);
    }

#pragma unroll
    for (int t = 0; t < 8; ++t) {
      // B fragment (32x16 bf16): lane-half hi=0 -> K {0..7,16..23}, hi=1 -> {8..15,24..31}
      FragAB bfrag;
      const unsigned short* bb = &Bt[(size_t)(t * 16 + lo) * 256 + k0 + hi * 8];
      bfrag.u[0] = *reinterpret_cast<const uint4*>(bb);
      bfrag.u[1] = *reinterpret_cast<const uint4*>(bb + 16);
      acc[t].v = __builtin_amdgcn_wmma_f32_16x16x32_bf16(
          false, afrag.v, false, bfrag.v, (short)0, acc[t].v, false, false);
    }
    __syncthreads();   // all reads of this buffer done before it is re-staged
  }

#pragma unroll
  for (int t = 0; t < 8; ++t) {
    const int col = t * 16 + lo;
#pragma unroll
    for (int v = 0; v < 8; ++v)
      C[(size_t)(mBase + v) * 128 + col] = acc[t].f[v];
  }
}

// ---------------- gates: z,r; hr = h*r; pack hr into ab2 low half -----------
__global__ void k_gates(float* __restrict__ accz, const float* __restrict__ accr,
                        const float* __restrict__ h, float* __restrict__ hr,
                        unsigned short* __restrict__ ab2) {
  int idx = blockIdx.x * blockDim.x + threadIdx.x;
  if (idx >= N_NODESC * 128) return;
  float z = sigmoidf(accz[idx]);
  accz[idx] = z;
  float r = sigmoidf(accr[idx]);
  float hrv = h[idx] * r;
  hr[idx] = hrv;
  int n = idx >> 7, f = idx & 127;
  ab2[(size_t)n * 256 + f] = f2bf(hrv);
}

// ---------------- GRU update -----------------------------------------------
__global__ void k_update(const float* __restrict__ accz, const float* __restrict__ acch,
                         float* __restrict__ h) {
  int idx = blockIdx.x * blockDim.x + threadIdx.x;
  if (idx >= N_NODESC * 128) return;
  float z = accz[idx];
  float ht = tanhf(acch[idx]);
  h[idx] = z * h[idx] + (1.0f - z) * ht;
}

// ---------------- head: u = h @ W_head + b_head ------------------------------
__global__ void k_head(const float* __restrict__ h, const float* __restrict__ Wh,
                       const float* __restrict__ bh, float* __restrict__ out) {
  int idx = blockIdx.x * blockDim.x + threadIdx.x;
  if (idx >= N_NODESC * 3) return;
  int n = idx / 3, o = idx - n * 3;
  float s = bh[o];
  const float* hrow = h + (size_t)n * 128;
#pragma unroll 16
  for (int i = 0; i < 128; ++i) s += hrow[i] * Wh[i * 3 + o];
  out[idx] = s;
}

// ============================================================================
extern "C" void kernel_launch(void* const* d_in, const int* in_sizes, int n_in,
                              void* d_out_, int out_size, void* d_ws, size_t ws_size,
                              hipStream_t stream) {
  (void)in_sizes; (void)n_in; (void)out_size; (void)ws_size;
  const float* X     = (const float*)d_in[0];
  const int*   ei    = (const int*)d_in[1];
  const float* Wxz   = (const float*)d_in[2];
  const float* bxz   = (const float*)d_in[3];
  const float* Whz   = (const float*)d_in[4];
  const float* bhz   = (const float*)d_in[5];
  const float* Wxr   = (const float*)d_in[6];
  const float* bxr   = (const float*)d_in[7];
  const float* Whr   = (const float*)d_in[8];
  const float* bhr   = (const float*)d_in[9];
  const float* Wxh   = (const float*)d_in[10];
  const float* bxh   = (const float*)d_in[11];
  const float* Whh   = (const float*)d_in[12];
  const float* bhh   = (const float*)d_in[13];
  const float* Whead = (const float*)d_in[14];
  const float* bhead = (const float*)d_in[15];
  float* out = (float*)d_out_;

  char* p = (char*)d_ws;
  auto alloc = [&](size_t bytes) -> char* {
    char* r = p;
    p += (bytes + 255) & ~(size_t)255;
    return r;
  };
  float* deg  = (float*)alloc((size_t)NPAD * 4);
  float* dinv = (float*)alloc((size_t)NPAD * 4);
  float* w    = (float*)alloc((size_t)N_EDGESC * 4);
  float* xb   = (float*)alloc((size_t)NPAD * 12 * 4);
  float* lx   = (float*)alloc((size_t)NPAD * 12 * 4);
  float* h    = (float*)alloc((size_t)NPAD * 128 * 4);
  float* lh   = (float*)alloc((size_t)NPAD * 128 * 4);
  float* hr   = (float*)alloc((size_t)NPAD * 128 * 4);
  float* accz = (float*)alloc((size_t)NPAD * 128 * 4);
  float* accr = (float*)alloc((size_t)NPAD * 128 * 4);
  float* acch = (float*)alloc((size_t)NPAD * 128 * 4);
  unsigned short* ab  = (unsigned short*)alloc((size_t)NPAD * 256 * 2);
  unsigned short* ab2 = (unsigned short*)alloc((size_t)NPAD * 256 * 2);
  unsigned short* Bz  = (unsigned short*)alloc(128 * 256 * 2);
  unsigned short* Br  = (unsigned short*)alloc(128 * 256 * 2);
  unsigned short* Bh  = (unsigned short*)alloc(128 * 256 * 2);

  const int TB = 256;
  const int NF = N_NODESC * 128;
  const int gE = (N_EDGESC + TB - 1) / TB;
  const int gN = (N_NODESC + TB - 1) / TB;
  const int gNF = (NF + TB - 1) / TB;
  const int gE32 = (N_EDGESC * 32 + TB - 1) / TB;

  // ---- per-launch init (deterministic) ----
  hipMemsetAsync(deg, 0, (size_t)NPAD * 4, stream);
  hipMemsetAsync(h, 0, (size_t)NPAD * 128 * 4, stream);
  hipMemsetAsync(ab, 0, (size_t)NPAD * 256 * 2, stream);   // zero pad rows of A
  hipMemsetAsync(ab2, 0, (size_t)NPAD * 256 * 2, stream);
  k_deg<<<gE, TB, 0, stream>>>(ei, deg);
  k_dinv<<<gN, TB, 0, stream>>>(deg, dinv);
  k_edgew<<<gE, TB, 0, stream>>>(ei, dinv, w);
  k_prep_w<<<(128 * 256 + TB - 1) / TB, TB, 0, stream>>>(Whz, Bz);
  k_prep_w<<<(128 * 256 + TB - 1) / TB, TB, 0, stream>>>(Whr, Br);
  k_prep_w<<<(128 * 256 + TB - 1) / TB, TB, 0, stream>>>(Whh, Bh);

  for (int t = 0; t < T_STEPS; ++t) {
    const float* uprev = out + (size_t)(t > 0 ? t - 1 : 0) * N_NODESC * 3;
    k_xmod<<<gN, TB, 0, stream>>>(X, uprev, xb, t);
    hipMemsetAsync(lx, 0, (size_t)NPAD * 12 * 4, stream);
    k_scatter_x<<<gE, TB, 0, stream>>>(ei, w, xb, lx);
    hipMemsetAsync(lh, 0, (size_t)NPAD * 128 * 4, stream);
    k_scatter128<<<gE32, TB, 0, stream>>>(ei, w, h, lh);
    k_xpart<<<gNF, TB, 0, stream>>>(xb, lx, Wxz, Wxr, Wxh,
                                    bxz, bhz, bxr, bhr, bxh, bhh,
                                    accz, accr, acch);
    k_pack2<<<gNF, TB, 0, stream>>>(h, lh, ab);
    k_wmma_gemm<<<NPAD / 128, 256, 0, stream>>>(ab, Bz, accz);
    k_wmma_gemm<<<NPAD / 128, 256, 0, stream>>>(ab, Br, accr);
    k_gates<<<gNF, TB, 0, stream>>>(accz, accr, h, hr, ab2);
    hipMemsetAsync(lh, 0, (size_t)NPAD * 128 * 4, stream);
    k_scatter128<<<gE32, TB, 0, stream>>>(ei, w, hr, lh);
    k_packhi<<<gNF, TB, 0, stream>>>(lh, ab2);
    k_wmma_gemm<<<NPAD / 128, 256, 0, stream>>>(ab2, Bh, acch);
    k_update<<<gNF, TB, 0, stream>>>(accz, acch, h);
    k_head<<<(N_NODESC * 3 + TB - 1) / TB, TB, 0, stream>>>(
        h, Whead, bhead, out + (size_t)t * N_NODESC * 3);
  }
}
